// SchNetInteraction_14783277433357
// MI455X (gfx1250) — matrix operand order repaired and checked by reading, not compiled, not run on hardware
//
#include <hip/hip_runtime.h>
#include <hip/hip_bf16.h>

// ---------------------------------------------------------------------------
// SchNet interaction block, fused for gfx1250 (wave32, WMMA f16 -> f32 acc).
//
// Pipeline:
//   prep:    weights f32 -> f16, pre-swizzled into WMMA B-fragment order
//   in2f:    y = x @ W_in2f                       [8192x128] WMMA GEMM
//   cfconv:  per (b,a): filter = (ssp(f_ij@W1+b1)@W2+b2)*cutoff*mask,
//            agg = sum_nbr filter * y[neighbor]    (fully fused, LDS-resident)
//   out:     v = ssp(agg@W_f2out+b)@W_dense+b      fused double GEMM
// ---------------------------------------------------------------------------

typedef __attribute__((ext_vector_type(16))) _Float16 v16h;
typedef __attribute__((ext_vector_type(8)))  float    v8f;

#define PI_OVER_CUTOFF 0.6283185307179586f
#define LN2F 0.6931471805599453f

__device__ __forceinline__ float ssp_f(float v) {
  // shifted softplus: softplus(v) - ln(2)
  float sp = (v > 20.f) ? v : log1pf(__expf(v));
  return sp - LN2F;
}

__device__ __forceinline__ v8f wmma_f16(v16h a, v16h b, v8f c) {
  return __builtin_amdgcn_wmma_f32_16x16x32_f16(
      /*neg_a=*/false, a, /*neg_b=*/false, b,
      /*c_mod=*/(short)0, c, /*reuse_a=*/false, /*reuse_b=*/false);
}

// A-matrix 16x32 f16 fragment from row-major LDS (ISA 7.12.2 layout):
// lane L holds row M = L&15; half h holds K = 16*(h>>3) + (h&7) + 8*(L>>4).
__device__ __forceinline__ v16h load_frag_A_lds(const _Float16* base, int ld,
                                                int row0, int k0, int lane) {
  const _Float16* rp = base + (row0 + (lane & 15)) * ld + k0 + ((lane >> 4) << 3);
  v16h a;
#pragma unroll
  for (int h = 0; h < 16; ++h) {
    a[h] = rp[((h >> 3) << 4) + (h & 7)];
  }
  return a;
}

// B-fragment (pre-swizzled in global): one contiguous v16h (32B) per lane.
__device__ __forceinline__ v16h load_frag_B(const _Float16* frag, int fidx, int lane) {
  return *(const v16h*)(frag + ((size_t)fidx << 9) + (lane << 4));
}

// ---------------------------------------------------------------------------
// Prep: f32 weight [Ksrc x N] -> f16 B-fragment order, K padded to Kpad.
// Fragment f = kb*(N/16)+cb holds the 32x16 block (K=kb*32.., N=cb*16..).
// B layout: lane L -> col N = L&15; half h -> K = 16*(h>>3) + (h&7) + 8*(L>>4).
// ---------------------------------------------------------------------------
__global__ void prep_frag_kernel(const float* __restrict__ src,
                                 _Float16* __restrict__ dst,
                                 int Ksrc, int Kpad, int N) {
  int i = blockIdx.x * blockDim.x + threadIdx.x;
  int total = Kpad * N;
  if (i >= total) return;
  int f = i >> 9;
  int within = i & 511;
  int L = within >> 4;
  int h = within & 15;
  int n16 = N >> 4;
  int kb = f / n16;
  int cb = f - kb * n16;
  int k = kb * 32 + ((h >> 3) << 4) + ((L >> 4) << 3) + (h & 7);
  int n = cb * 16 + (L & 15);
  float v = (k < Ksrc) ? src[k * N + n] : 0.f;
  dst[i] = (_Float16)v;
}

// ---------------------------------------------------------------------------
// in2f: y[8192x128] = x[8192x128] @ W_in2f[128x128]
// 64 blocks x 256 threads (8 waves); wave w owns rows 16w..16w+15 of the tile.
// ---------------------------------------------------------------------------
__global__ __launch_bounds__(256) void schnet_in2f_kernel(
    const float* __restrict__ x, const _Float16* __restrict__ wfrag,
    float* __restrict__ y) {
  __shared__ _Float16 sX[128 * 128];
  const int tid = threadIdx.x;
  const int row0 = blockIdx.x * 128;
  for (int i = tid; i < 128 * 128; i += 256)
    sX[i] = (_Float16)x[(size_t)(row0 + (i >> 7)) * 128 + (i & 127)];
  __syncthreads();

  const int wv = tid >> 5, lane = tid & 31;
  const int c_lo = lane & 15;
  const int rbase = wv * 16 + ((lane >> 4) << 3);
  for (int cb = 0; cb < 8; ++cb) {
    v8f acc = {};
#pragma unroll
    for (int kb = 0; kb < 4; ++kb) {
      v16h a = load_frag_A_lds(sX, 128, wv * 16, kb * 32, lane);
      v16h b = load_frag_B(wfrag, (kb << 3) + cb, lane);
      acc = wmma_f16(a, b, acc);
    }
    int c = cb * 16 + c_lo;
#pragma unroll
    for (int j = 0; j < 8; ++j)
      y[(size_t)(row0 + rbase + j) * 128 + c] = acc[j];
  }
}

// ---------------------------------------------------------------------------
// cfconv: one block per (b,a) pair. 128 threads = 4 waves; wave w owns the
// 16-row slab 16w of the (padded) 64-row neighbor dimension.
// Fuses: GEMM1 -> ssp -> GEMM2 -> bias -> cutoff*mask -> *y[nbr] -> reduce.
// ---------------------------------------------------------------------------
__global__ __launch_bounds__(128) void schnet_cfconv_kernel(
    const float* __restrict__ fij, const float* __restrict__ rij,
    const float* __restrict__ nmask, const int* __restrict__ nbrs,
    const _Float16* __restrict__ w1frag, const float* __restrict__ b1,
    const _Float16* __restrict__ w2frag, const float* __restrict__ b2,
    const float* __restrict__ y, float* __restrict__ agg) {
  __shared__ _Float16 sA[64 * 64];    // f_ij tile (48x50 padded), f16
  __shared__ _Float16 sH[64 * 128];   // ssp(GEMM1) intermediate, f16
  __shared__ float    sY[48 * 128];   // gathered neighbor features
  __shared__ float    sC[64];         // cutoff * mask per neighbor
  __shared__ float    sAgg[128];      // aggregation accumulator

  const int p = blockIdx.x;           // pair index = b*512 + a
  const int bidx = p >> 9;
  const int tid = threadIdx.x;
  const float* fbase = fij + (size_t)p * 48 * 50;

  // stage f_ij -> f16, zero-padded to 64x64
  for (int i = tid; i < 64 * 64; i += 128) {
    int r = i >> 6, k = i & 63;
    sA[i] = (r < 48 && k < 50) ? (_Float16)fbase[r * 50 + k] : (_Float16)0.f;
  }
  // cosine cutoff * neighbor mask
  if (tid < 64) {
    float cv = 0.f;
    if (tid < 48) {
      float r = rij[(size_t)p * 48 + tid];
      float m = nmask[(size_t)p * 48 + tid];
      cv = (r < 5.0f) ? 0.5f * (__cosf(r * PI_OVER_CUTOFF) + 1.f) * m : 0.f;
    }
    sC[tid] = cv;
  }
  sAgg[tid] = 0.f;
  // gather y rows for the 48 neighbors (coalesced per row)
  for (int n = 0; n < 48; ++n) {
    int nb = nbrs[(size_t)p * 48 + n];
    sY[n * 128 + tid] = y[((size_t)(bidx * 512 + nb)) * 128 + tid];
  }
  __syncthreads();

  const int wv = tid >> 5, lane = tid & 31;
  const int c_lo = lane & 15;
  const int rbase = wv * 16 + ((lane >> 4) << 3);

  // GEMM1: H = ssp(A @ W1 + b1), K = 64 (padded from 50)
  for (int cb = 0; cb < 8; ++cb) {
    v8f acc = {};
#pragma unroll
    for (int kb = 0; kb < 2; ++kb) {
      v16h a = load_frag_A_lds(sA, 64, wv * 16, kb * 32, lane);
      v16h b = load_frag_B(w1frag, (kb << 3) + cb, lane);
      acc = wmma_f16(a, b, acc);
    }
    int c = cb * 16 + c_lo;
    float bias = b1[c];
#pragma unroll
    for (int j = 0; j < 8; ++j) {
      int r = rbase + j;
      float hv = (r < 48) ? ssp_f(acc[j] + bias) : 0.f;
      sH[r * 128 + c] = (_Float16)hv;
    }
  }
  __syncthreads();

  // GEMM2 + cutoff + gather-multiply + neighbor reduction
  for (int cb = 0; cb < 8; ++cb) {
    v8f acc = {};
#pragma unroll
    for (int kb = 0; kb < 4; ++kb) {
      v16h a = load_frag_A_lds(sH, 128, wv * 16, kb * 32, lane);
      v16h b = load_frag_B(w2frag, (kb << 3) + cb, lane);
      acc = wmma_f16(a, b, acc);
    }
    int c = cb * 16 + c_lo;
    float bias = b2[c];
    float psum = 0.f;
#pragma unroll
    for (int j = 0; j < 8; ++j) {
      int r = rbase + j;
      if (r < 48) psum += (acc[j] + bias) * sC[r] * sY[r * 128 + c];
    }
    atomicAdd(&sAgg[c], psum);   // ds_add_f32: 8 contributors per column
  }
  __syncthreads();
  agg[(size_t)p * 128 + tid] = sAgg[tid];
}

// ---------------------------------------------------------------------------
// out: v = ssp(agg @ W_f2out + b_f2out) @ W_dense + b_dense
// Fused double GEMM over 128-row tiles; one 32KB LDS tile reused.
// ---------------------------------------------------------------------------
__global__ __launch_bounds__(256) void schnet_out_kernel(
    const float* __restrict__ agg,
    const _Float16* __restrict__ w1frag, const float* __restrict__ b1,
    const _Float16* __restrict__ w2frag, const float* __restrict__ b2,
    float* __restrict__ out) {
  __shared__ _Float16 sX[128 * 128];
  const int tid = threadIdx.x;
  const int row0 = blockIdx.x * 128;
  for (int i = tid; i < 128 * 128; i += 256)
    sX[i] = (_Float16)agg[(size_t)(row0 + (i >> 7)) * 128 + (i & 127)];
  __syncthreads();

  const int wv = tid >> 5, lane = tid & 31;
  const int c_lo = lane & 15;
  const int rbase = wv * 16 + ((lane >> 4) << 3);

  float res[8][8];
  for (int cb = 0; cb < 8; ++cb) {
    v8f acc = {};
#pragma unroll
    for (int kb = 0; kb < 4; ++kb) {
      v16h a = load_frag_A_lds(sX, 128, wv * 16, kb * 32, lane);
      v16h b = load_frag_B(w1frag, (kb << 3) + cb, lane);
      acc = wmma_f16(a, b, acc);
    }
    float bias = b1[cb * 16 + c_lo];
#pragma unroll
    for (int j = 0; j < 8; ++j) res[cb][j] = ssp_f(acc[j] + bias);
  }
  __syncthreads();   // all waves done reading sX
  for (int cb = 0; cb < 8; ++cb) {
    int c = cb * 16 + c_lo;
#pragma unroll
    for (int j = 0; j < 8; ++j)
      sX[(rbase + j) * 128 + c] = (_Float16)res[cb][j];
  }
  __syncthreads();

  for (int cb = 0; cb < 8; ++cb) {
    v8f acc = {};
#pragma unroll
    for (int kb = 0; kb < 4; ++kb) {
      v16h a = load_frag_A_lds(sX, 128, wv * 16, kb * 32, lane);
      v16h b = load_frag_B(w2frag, (kb << 3) + cb, lane);
      acc = wmma_f16(a, b, acc);
    }
    int c = cb * 16 + c_lo;
    float bias = b2[c];
#pragma unroll
    for (int j = 0; j < 8; ++j)
      out[(size_t)(row0 + rbase + j) * 128 + c] = acc[j] + bias;
  }
}

// ---------------------------------------------------------------------------
extern "C" void kernel_launch(void* const* d_in, const int* in_sizes, int n_in,
                              void* d_out, int out_size, void* d_ws, size_t ws_size,
                              hipStream_t stream) {
  (void)in_sizes; (void)n_in; (void)out_size; (void)ws_size;
  const float* x       = (const float*)d_in[0];   // [16,512,128]
  const float* r_ij    = (const float*)d_in[1];   // [16,512,48]
  const float* f_ij    = (const float*)d_in[2];   // [16,512,48,50]
  const float* nmask   = (const float*)d_in[3];   // [16,512,48]
  const float* W_f1    = (const float*)d_in[4];   // [50,128]
  const float* b_f1    = (const float*)d_in[5];   // [128]
  const float* W_f2    = (const float*)d_in[6];   // [128,128]
  const float* b_f2    = (const float*)d_in[7];   // [128]
  const float* W_in2f  = (const float*)d_in[8];   // [128,128]
  const float* W_f2out = (const float*)d_in[9];   // [128,128]
  const float* b_f2out = (const float*)d_in[10];  // [128]
  const float* W_dense = (const float*)d_in[11];  // [128,128]
  const float* b_dense = (const float*)d_in[12];  // [128]
  const int*   nbrs    = (const int*)d_in[13];    // [16,512,48]
  float* out = (float*)d_out;                      // [16,512,128]

  // workspace layout
  char* ws = (char*)d_ws;
  _Float16* w1frag     = (_Float16*)ws;             //  64*128 halves
  _Float16* w2frag     = w1frag + 64 * 128;         // 128*128
  _Float16* win2frag   = w2frag + 128 * 128;
  _Float16* wf2outfrag = win2frag + 128 * 128;
  _Float16* wdensefrag = wf2outfrag + 128 * 128;
  float* yBuf   = (float*)(ws + 147456);            // 8192*128 f32
  float* aggBuf = yBuf + 8192 * 128;                // 8192*128 f32

  // weight pre-swizzle to f16 B-fragment order
  prep_frag_kernel<<<(64 * 128 + 255) / 256, 256, 0, stream>>>(W_f1, w1frag, 50, 64, 128);
  prep_frag_kernel<<<(128 * 128 + 255) / 256, 256, 0, stream>>>(W_f2, w2frag, 128, 128, 128);
  prep_frag_kernel<<<(128 * 128 + 255) / 256, 256, 0, stream>>>(W_in2f, win2frag, 128, 128, 128);
  prep_frag_kernel<<<(128 * 128 + 255) / 256, 256, 0, stream>>>(W_f2out, wf2outfrag, 128, 128, 128);
  prep_frag_kernel<<<(128 * 128 + 255) / 256, 256, 0, stream>>>(W_dense, wdensefrag, 128, 128, 128);

  // y = x @ W_in2f
  schnet_in2f_kernel<<<64, 256, 0, stream>>>(x, win2frag, yBuf);
  // fused filter network + cfconv aggregation, one block per (b,a)
  schnet_cfconv_kernel<<<16 * 512, 128, 0, stream>>>(
      f_ij, r_ij, nmask, nbrs, w1frag, b_f1, w2frag, b_f2, yBuf, aggBuf);
  // v = ssp(agg @ W_f2out + b) @ W_dense + b
  schnet_out_kernel<<<64, 256, 0, stream>>>(
      aggBuf, wf2outfrag, b_f2out, wdensefrag, b_dense, out);
}
